// ParallelDecoder_37563783970971
// MI455X (gfx1250) — compile-verified, matrix operand order checked by reading, not run on hardware
//
#include <hip/hip_runtime.h>
#include <hip/hip_bf16.h>
#include <math.h>

// ---------------------------------------------------------------------------
// Problem constants (from the reference)
// ---------------------------------------------------------------------------
#define BN   512
#define SDIM 49
#define SSGD 49
#define LDIM 20
#define DDIM 512
#define HH   8
#define DKD  64
#define MMEM 40
#define VV   1024
#define HDK  512     // H*DK
#define TSTEPS 19    // L-1
#define LP_PAD 128   // l_param 100 -> padded to 128 so every GEMM dim is aligned

typedef __attribute__((ext_vector_type(16))) __bf16 v16bf;
typedef __attribute__((ext_vector_type(8)))  float  v8f;

// ---------------------------------------------------------------------------
// Weight pre-pack: rewrite a B operand (fp32 [K,N] row-major, or with transB a
// row-major [N,K] used as B^T) into bf16 in exact WMMA B-fragment order:
//   Bp[ chunk ][ tileN ][ lane ][ e ]  with
//   k = chunk*32 + ((e>>3)<<4) + ((lane>>4)<<3) + (e&7),  n = tileN*16 + (lane&15)
// Out-of-range (k>=realK or n>=realN) elements are zero-filled (l_param pad).
// Weights are step-invariant: packed once, then re-read 19x from L2.
// ---------------------------------------------------------------------------
__global__ void pack_b_kernel(const float* __restrict__ W, __bf16* __restrict__ Bp,
                              int Np, long ldw, int realK, int realN, int transB)
{
    const size_t idx  = (size_t)blockIdx.x * 256 + threadIdx.x;
    const int e       = (int)(idx & 15);
    const int lane    = (int)((idx >> 4) & 31);
    const size_t rest = idx >> 9;
    const int nt      = Np >> 4;
    const int tileN   = (int)(rest % nt);
    const int chunk   = (int)(rest / nt);
    const int k = chunk * 32 + ((e >> 3) << 4) + ((lane >> 4) << 3) + (e & 7);
    const int n = tileN * 16 + (lane & 15);
    float v = 0.f;
    if (k < realK && n < realN)
        v = transB ? W[(size_t)n * ldw + k] : W[(size_t)k * ldw + n];
    Bp[idx] = (__bf16)v;
}

// ---------------------------------------------------------------------------
// WMMA GEMM: C[M,N] = act( A[M,K] @ Bpacked + (ACC ? C : 0) + bias )
// Requirements guaranteed by the caller: M%16==0, K%32==0, and
//   NT==4: N%512==0 (8 waves x 4 tiles = 512 cols/block)
//   NT==1: N%128==0
// Each wave computes a 16x(16*NT) strip: one A fragment (4x float4 loads +
// 8 cvt_pk_bf16) is reused by NT consecutive B fragments (each one contiguous
// 32B load of pre-packed bf16) -> NT back-to-back v_wmma_f32_16x16x32_bf16.
// fp32 accumulation; ISA 7.12.2 wave32 layouts.
// ---------------------------------------------------------------------------
template<int ACT, bool ACC, int NT>
__global__ __launch_bounds__(256) void wmma_gemm_kernel(
    const float* __restrict__ A, long lda,
    const __bf16* __restrict__ Bp,
    const float* __restrict__ bias,
    float* __restrict__ C,
    int Ndim, int Kdim)
{
    const int lane  = threadIdx.x;                 // 0..31
    const int wv    = threadIdx.y;                 // 0..7
    const int tn0   = (blockIdx.x * 8 + wv) * NT;  // first 16-col tile of strip
    const int tileM = blockIdx.y;
    const int half  = lane >> 4;                   // 0: lanes 0-15, 1: lanes 16-31
    const int rowA  = tileM * 16 + (lane & 15);

    const float4* ap = (const float4*)(A + (size_t)rowA * lda + half * 8);
    const v16bf*  bp = (const v16bf*)(Bp) + ((size_t)tn0 * 32 + lane);
    const size_t  bstride = (size_t)(Ndim >> 4) * 32;   // v16bf per K-chunk
    const int nChunks = Kdim >> 5;

    v8f acc[NT];
#pragma unroll
    for (int i = 0; i < NT; ++i) acc[i] = (v8f){};

    for (int c = 0; c < nChunks; ++c) {
        const float4* a4 = ap + (size_t)c * 8;    // 32 floats per chunk
        const float4 a0 = a4[0];                  // k = c*32 + half*8 + 0..3
        const float4 a1 = a4[1];                  // .. + 4..7
        const float4 a2 = a4[4];                  // k = c*32 + 16 + half*8 + 0..3
        const float4 a3 = a4[5];                  // .. + 4..7
        v16bf af;
        af[0]=(__bf16)a0.x;  af[1]=(__bf16)a0.y;  af[2]=(__bf16)a0.z;  af[3]=(__bf16)a0.w;
        af[4]=(__bf16)a1.x;  af[5]=(__bf16)a1.y;  af[6]=(__bf16)a1.z;  af[7]=(__bf16)a1.w;
        af[8]=(__bf16)a2.x;  af[9]=(__bf16)a2.y;  af[10]=(__bf16)a2.z; af[11]=(__bf16)a2.w;
        af[12]=(__bf16)a3.x; af[13]=(__bf16)a3.y; af[14]=(__bf16)a3.z; af[15]=(__bf16)a3.w;
        const v16bf* bc = bp + (size_t)c * bstride;
#pragma unroll
        for (int i = 0; i < NT; ++i) {
            const v16bf bf = bc[(size_t)i * 32];  // next 16-col tile: +1KB
            acc[i] = __builtin_amdgcn_wmma_f32_16x16x32_bf16(
                         false, af, false, bf, (short)0, acc[i], false, false);
        }
    }

#pragma unroll
    for (int i = 0; i < NT; ++i) {
        const int colB = (tn0 + i) * 16 + (lane & 15);
        const float b0 = bias ? bias[colB] : 0.f;
        float* cp = C + ((size_t)tileM * 16 + (half << 3)) * Ndim + colB;
#pragma unroll
        for (int r = 0; r < 8; ++r) {             // C/D layout: VGPR r -> M = r + 8*half
            float v = acc[i][r] + b0;
            if (ACC) v += cp[(size_t)r * Ndim];
            if (ACT == 1) v = fmaxf(v, 0.f);
            cp[(size_t)r * Ndim] = v;
        }
    }
}

static void launch_gemm(hipStream_t st, const float* A, long lda, const __bf16* Bp,
                        const float* bias, float* C, int M, int N, int K,
                        int act, bool accF)
{
    dim3 blk(32, 8);
    if ((N & 511) == 0) {                         // N = 512/1024/2048: 4-tile strips
        dim3 grd(N / 512, M / 16);
        if (!accF && act == 0)
            wmma_gemm_kernel<0,false,4><<<grd,blk,0,st>>>(A,lda,Bp,bias,C,N,K);
        else if (!accF && act == 1)
            wmma_gemm_kernel<1,false,4><<<grd,blk,0,st>>>(A,lda,Bp,bias,C,N,K);
        else if (accF && act == 0)
            wmma_gemm_kernel<0,true,4><<<grd,blk,0,st>>>(A,lda,Bp,bias,C,N,K);
        else
            wmma_gemm_kernel<1,true,4><<<grd,blk,0,st>>>(A,lda,Bp,bias,C,N,K);
    } else {                                      // N = 128 (padded l_param)
        dim3 grd(N / 128, M / 16);
        if (!accF && act == 0)
            wmma_gemm_kernel<0,false,1><<<grd,blk,0,st>>>(A,lda,Bp,bias,C,N,K);
        else if (!accF && act == 1)
            wmma_gemm_kernel<1,false,1><<<grd,blk,0,st>>>(A,lda,Bp,bias,C,N,K);
        else if (accF && act == 0)
            wmma_gemm_kernel<0,true,1><<<grd,blk,0,st>>>(A,lda,Bp,bias,C,N,K);
        else
            wmma_gemm_kernel<1,true,1><<<grd,blk,0,st>>>(A,lda,Bp,bias,C,N,K);
    }
}

// ---------------------------------------------------------------------------
// Memory-augmented MHA: one wave per (b,h). Keys/values: Lk projected rows
// plus MMEM never-masked memory rows. wave32 shfl reductions for softmax.
// ---------------------------------------------------------------------------
__global__ __launch_bounds__(256) void mha_kernel(
    const float* __restrict__ Q, const float* __restrict__ Kp,
    const float* __restrict__ Vp, const float* __restrict__ mk,
    const float* __restrict__ mv, const unsigned char* __restrict__ mask,
    float* __restrict__ out, int Lk)
{
    const int Lt = Lk + MMEM;
    const int lane = threadIdx.x, wv = threadIdx.y;
    const int idx = blockIdx.x * 8 + wv;
    const int b = idx >> 3, h = idx & 7;

    __shared__ float sq[8][64];
    __shared__ float satt[8][128];

    const float* qp = Q + (size_t)b * HDK + h * DKD;
    sq[wv][lane] = qp[lane];
    sq[wv][lane + 32] = qp[lane + 32];
    __syncthreads();

    float sc[4];
    float mx = -1e30f;
    const int nIt = (Lt + 31) >> 5;
    for (int t = 0; t < nIt; ++t) {
        const int j = t * 32 + lane;
        float s = -1e30f;
        if (j < Lt) {
            const float* kr = (j < Lk) ? (Kp + ((size_t)b * Lk + j) * HDK + h * DKD)
                                       : (mk + (size_t)(j - Lk) * HDK + h * DKD);
            float a = 0.f;
#pragma unroll
            for (int d = 0; d < 64; ++d) a += sq[wv][d] * kr[d];
            s = a * 0.125f;                               // 1/sqrt(DK)
            if (j < Lk && mask[(size_t)b * Lk + j]) s = -1e9f;
        }
        sc[t] = s;
        mx = fmaxf(mx, s);
    }
    for (int o = 16; o; o >>= 1) mx = fmaxf(mx, __shfl_xor(mx, o));

    float sum = 0.f;
    for (int t = 0; t < nIt; ++t) {
        const int j = t * 32 + lane;
        float e = (j < Lt) ? expf(sc[t] - mx) : 0.f;
        sum += e;
        if (j < 128) satt[wv][j] = e;
    }
    for (int o = 16; o; o >>= 1) sum += __shfl_xor(sum, o);
    const float inv = 1.f / sum;
    __syncthreads();

    float o0 = 0.f, o1 = 0.f;
    for (int j = 0; j < Lt; ++j) {
        const float a = satt[wv][j] * inv;
        const float* vr = (j < Lk) ? (Vp + ((size_t)b * Lk + j) * HDK + h * DKD)
                                   : (mv + (size_t)(j - Lk) * HDK + h * DKD);
        o0 += a * vr[lane];
        o1 += a * vr[lane + 32];
    }
    out[(size_t)b * HDK + h * DKD + lane] = o0;
    out[(size_t)b * HDK + h * DKD + lane + 32] = o1;
}

// ---------------------------------------------------------------------------
// LayerNorm(mha_out + x) over D=512; x has row stride xstride (caption slice).
// ---------------------------------------------------------------------------
__global__ __launch_bounds__(256) void ln_res_kernel(
    const float* __restrict__ a, const float* __restrict__ xres, long xstride,
    float* __restrict__ out)
{
    const int b = blockIdx.x, tid = threadIdx.x;
    __shared__ float red[256];
    float v0 = a[(size_t)b * DDIM + tid]        + xres[(size_t)b * xstride + tid];
    float v1 = a[(size_t)b * DDIM + tid + 256]  + xres[(size_t)b * xstride + tid + 256];
    red[tid] = v0 + v1;
    __syncthreads();
    for (int o = 128; o; o >>= 1) { if (tid < o) red[tid] += red[tid + o]; __syncthreads(); }
    const float mean = red[0] * (1.f / 512.f);
    __syncthreads();
    const float d0 = v0 - mean, d1 = v1 - mean;
    red[tid] = d0 * d0 + d1 * d1;
    __syncthreads();
    for (int o = 128; o; o >>= 1) { if (tid < o) red[tid] += red[tid + o]; __syncthreads(); }
    const float inv = rsqrtf(red[0] * (1.f / 512.f) + 1e-5f);
    out[(size_t)b * DDIM + tid]       = d0 * inv;
    out[(size_t)b * DDIM + tid + 256] = d1 * inv;
}

// ---------------------------------------------------------------------------
// Bahdanau attention: e_s = tanh(encW1[b,s,:] + q[b,:]) . att_v ; softmax over S;
// context = sum_s a_s * enc[b,s,:].  One block per b, 8 waves split S.
// ---------------------------------------------------------------------------
__global__ __launch_bounds__(256) void addatt_kernel(
    const float* __restrict__ encW1, const float* __restrict__ q,
    const float* __restrict__ vw, const float* __restrict__ enc,
    const unsigned char* __restrict__ mask, float* __restrict__ vctx)
{
    const int b = blockIdx.x, tid = threadIdx.x;
    const int lane = tid & 31, wv = tid >> 5;
    __shared__ float a_sh[64];
    for (int s = wv; s < SDIM; s += 8) {
        float p = 0.f;
        for (int d = lane; d < DDIM; d += 32)
            p += tanhf(encW1[((size_t)b * SDIM + s) * DDIM + d] + q[(size_t)b * DDIM + d]) * vw[d];
        for (int o = 16; o; o >>= 1) p += __shfl_xor(p, o);
        if (lane == 0)
            a_sh[s] = mask[(size_t)b * SDIM + s] ? -1e9f : p;
    }
    __syncthreads();
    if (tid == 0) {
        float mx = -1e30f;
        for (int s = 0; s < SDIM; ++s) mx = fmaxf(mx, a_sh[s]);
        float sum = 0.f;
        for (int s = 0; s < SDIM; ++s) { float e = expf(a_sh[s] - mx); a_sh[s] = e; sum += e; }
        const float inv = 1.f / sum;
        for (int s = 0; s < SDIM; ++s) a_sh[s] *= inv;
    }
    __syncthreads();
    for (int d = tid; d < DDIM; d += 256) {
        float acc = 0.f;
        for (int s = 0; s < SDIM; ++s)
            acc += a_sh[s] * enc[((size_t)b * SDIM + s) * DDIM + d];
        vctx[(size_t)b * DDIM + d] = acc;
    }
}

// ---------------------------------------------------------------------------
// Elementwise kernels
// ---------------------------------------------------------------------------
__device__ __forceinline__ float sigm(float x) { return 1.f / (1.f + expf(-x)); }

__global__ void lstm_ew_kernel(const float* __restrict__ gates,
                               const float* __restrict__ cprev,
                               float* __restrict__ h, float* __restrict__ c)
{
    const int idx = blockIdx.x * 256 + threadIdx.x;     // B*D
    const int b = idx >> 9, d = idx & 511;
    const float* g = gates + (size_t)b * 2048;
    const float gi = g[d], gf = g[512 + d], gg = g[1024 + d], go = g[1536 + d];
    const float c2 = sigm(gf) * cprev[idx] + sigm(gi) * tanhf(gg);
    c[idx] = c2;
    h[idx] = sigm(go) * tanhf(c2);
}

__global__ void nwbuild_kernel(const float* __restrict__ mean, const float* __restrict__ h2,
                               const float* __restrict__ h3, const float* __restrict__ x1,
                               float* __restrict__ nw)
{
    const int idx = blockIdx.x * 256 + threadIdx.x;
    const int b = idx >> 9, d = idx & 511;
    nw[(size_t)b * 1024 + d]       = (mean[idx] + h2[idx] + h3[idx]) * (1.f / 3.f);
    nw[(size_t)b * 1024 + 512 + d] = x1[idx];
}

__global__ void concat_kernel(const float* __restrict__ a, const float* __restrict__ aadd,
                              const float* __restrict__ bsrc, float* __restrict__ dst)
{
    const int idx = blockIdx.x * 256 + threadIdx.x;
    const int b = idx >> 9, d = idx & 511;
    float v = a[idx];
    if (aadd) v += aadd[idx];
    dst[(size_t)b * 1024 + d]       = v;
    dst[(size_t)b * 1024 + 512 + d] = bsrc[idx];
}

__global__ void reluadd_kernel(const float* __restrict__ a, const float* __restrict__ b,
                               float* __restrict__ out)
{
    const int idx = blockIdx.x * 256 + threadIdx.x;
    out[idx] = fmaxf(a[idx], 0.f) + b[idx];
}

__global__ void dup_kernel(const float* __restrict__ src, float* __restrict__ d1,
                           float* __restrict__ d2)
{
    const int idx = blockIdx.x * 256 + threadIdx.x;
    const float v = src[idx];
    d1[idx] = v;
    if (d2) d2[idx] = v;
}

// t100 rows are padded to LP_PAD=128; cols 100..127 are zero and stay zero.
__global__ void softmax100_kernel(float* __restrict__ x)
{
    const int b = blockIdx.x, lane = threadIdx.x;
    float v[4];
    float mx = -1e30f;
#pragma unroll
    for (int i = 0; i < 4; ++i) {
        const int j = lane + i * 32;
        v[i] = (j < 100) ? x[(size_t)b * LP_PAD + j] : -1e30f;
        mx = fmaxf(mx, v[i]);
    }
    for (int o = 16; o; o >>= 1) mx = fmaxf(mx, __shfl_xor(mx, o));
    float s = 0.f;
#pragma unroll
    for (int i = 0; i < 4; ++i) {
        const int j = lane + i * 32;
        v[i] = (j < 100) ? expf(v[i] - mx) : 0.f;
        s += v[i];
    }
    for (int o = 16; o; o >>= 1) s += __shfl_xor(s, o);
    const float inv = 1.f / s;
#pragma unroll
    for (int i = 0; i < 4; ++i) {
        const int j = lane + i * 32;
        if (j < 100) x[(size_t)b * LP_PAD + j] = v[i] * inv;
    }
}

__global__ void combine_kernel(const float* __restrict__ r1, const float* __restrict__ r2,
                               const float* __restrict__ r3, const float* __restrict__ r4,
                               const float* __restrict__ h2new, float* __restrict__ h3out,
                               float* __restrict__ ys, int t)
{
    const int idx = blockIdx.x * 256 + threadIdx.x;
    const int b = idx >> 9, d = idx & 511;
    const float it = r1[idx] + r2[idx];
    const float gt = sigm(r3[idx] + r4[idx]);
    const float h3 = gt * it;
    h3out[idx] = h3;
    const size_t base = ((size_t)b * TSTEPS + t) * 1024;
    ys[base + d]       = h2new[idx];
    ys[base + 512 + d] = h3;
}

__global__ void fillout_kernel(const float* __restrict__ ys, float* __restrict__ outb)
{
    const size_t idx = (size_t)blockIdx.x * 256 + threadIdx.x;  // B*20*1024
    const int d = (int)(idx & 1023);
    const size_t r = idx >> 10;
    const int row = (int)(r % 20);
    const int b   = (int)(r / 20);
    outb[idx] = (row == 0) ? 2.0f
                           : ys[((size_t)b * TSTEPS + (row - 1)) * 1024 + d];
}

__global__ void logsoftmax_kernel(const float* __restrict__ logits, float* __restrict__ out)
{
    const int row = blockIdx.x, tid = threadIdx.x;   // V=1024, 256 threads
    __shared__ float red[256];
    float v[4];
    float mx = -1e30f;
#pragma unroll
    for (int i = 0; i < 4; ++i) {
        v[i] = logits[(size_t)row * VV + tid + i * 256];
        mx = fmaxf(mx, v[i]);
    }
    red[tid] = mx;
    __syncthreads();
    for (int o = 128; o; o >>= 1) { if (tid < o) red[tid] = fmaxf(red[tid], red[tid + o]); __syncthreads(); }
    mx = red[0];
    __syncthreads();
    float s = 0.f;
#pragma unroll
    for (int i = 0; i < 4; ++i) s += expf(v[i] - mx);
    red[tid] = s;
    __syncthreads();
    for (int o = 128; o; o >>= 1) { if (tid < o) red[tid] += red[tid + o]; __syncthreads(); }
    const float lse = mx + logf(red[0]);
#pragma unroll
    for (int i = 0; i < 4; ++i)
        out[(size_t)row * VV + tid + i * 256] = v[i] - lse;
}

__global__ void argmax_kernel(const float* __restrict__ logits, int* __restrict__ out)
{
    const int idx = blockIdx.x * 256 + threadIdx.x;   // B*V
    const int b = idx >> 10, v = idx & 1023;
    float best = logits[((size_t)b * 20) * VV + v];
    int bi = 0;
    for (int t = 1; t < 20; ++t) {
        const float x = logits[((size_t)b * 20 + t) * VV + v];
        if (x > best) { best = x; bi = t; }            // first max kept (jnp semantics)
    }
    out[idx] = bi;
}

// ---------------------------------------------------------------------------
// Host orchestration
// ---------------------------------------------------------------------------
extern "C" void kernel_launch(void* const* d_in, const int* in_sizes, int n_in,
                              void* d_out, int out_size, void* d_ws, size_t ws_size,
                              hipStream_t stream)
{
    (void)in_sizes; (void)n_in; (void)out_size; (void)ws_size;

    // Inputs in setup_inputs() order
    const float* mean    = (const float*)d_in[0];
    const float* enc     = (const float*)d_in[1];
    /* d_in[2] caption2i unused by the reference math */
    const float* capv    = (const float*)d_in[3];
    const unsigned char* sgmask  = (const unsigned char*)d_in[4];   // bool [B,S]
    const float* sg      = (const float*)d_in[5];
    const unsigned char* sgmask2 = (const unsigned char*)d_in[6];   // bool [B,SSG]
    const float* init_h_w = (const float*)d_in[7];
    const float* init_h_b = (const float*)d_in[8];
    const float* init_c_w = (const float*)d_in[9];
    const float* init_c_b = (const float*)d_in[10];
    const float* dWq = (const float*)d_in[11];
    const float* dWk = (const float*)d_in[12];
    const float* dWv = (const float*)d_in[13];
    const float* dWo = (const float*)d_in[14];
    const float* d_mk = (const float*)d_in[15];
    const float* d_mv = (const float*)d_in[16];
    const float* l1_wih = (const float*)d_in[17];
    const float* l1_whh = (const float*)d_in[18];
    const float* l1_b   = (const float*)d_in[19];
    const float* l2_wih = (const float*)d_in[20];
    const float* l2_whh = (const float*)d_in[21];
    const float* l2_b   = (const float*)d_in[22];
    const float* att_w1 = (const float*)d_in[23];
    const float* att_w2 = (const float*)d_in[24];
    const float* att_v  = (const float*)d_in[25];
    const float* lWq = (const float*)d_in[26];
    const float* lWk = (const float*)d_in[27];
    const float* lWv = (const float*)d_in[28];
    const float* lWo = (const float*)d_in[29];
    const float* l_mk = (const float*)d_in[30];
    const float* l_mv = (const float*)d_in[31];
    const float* l_param = (const float*)d_in[32];
    const float* lin_w = (const float*)d_in[33];
    const float* lin_b = (const float*)d_in[34];
    const float* lin1_w = (const float*)d_in[35];
    const float* lin1_b = (const float*)d_in[36];
    const float* lin2_w = (const float*)d_in[37];
    const float* lin2_b = (const float*)d_in[38];
    const float* lin3_w = (const float*)d_in[39];
    const float* lin3_b = (const float*)d_in[40];
    const float* vocab_w = (const float*)d_in[41];

    // Workspace carve (deterministic layout each call)
    char* wp = (char*)d_ws;
    auto allocf = [&](size_t nfloats) -> float* {
        float* p = (float*)wp;
        wp += nfloats * sizeof(float);
        return p;
    };
    auto allocb = [&](size_t nelems) -> __bf16* {
        __bf16* p = (__bf16*)wp;
        wp += nelems * sizeof(__bf16);
        return p;
    };

    const size_t PROJ = (size_t)BN * SDIM * HDK;  // == B*SSG*HDK
    float* sgK   = allocf(PROJ);
    float* sgV   = allocf(PROJ);
    float* encK  = allocf(PROJ);
    float* encV  = allocf(PROJ);
    float* encW1 = allocf(PROJ);
    const size_t BD = (size_t)BN * DDIM;
    float* hA[3] = { allocf(BD), allocf(BD), allocf(BD) };
    float* hB[3] = { allocf(BD), allocf(BD), allocf(BD) };
    float* cA[2] = { allocf(BD), allocf(BD) };
    float* cB[2] = { allocf(BD), allocf(BD) };
    float* x1      = allocf(BD);
    float* qbuf    = allocf(BD);
    float* attout  = allocf(BD);
    float* mhaout  = allocf(BD);
    float* nw      = allocf(BD * 2);
    float* gates   = allocf((size_t)BN * 2048);
    float* t100    = allocf((size_t)BN * LP_PAD);
    float* h1npre  = allocf(BD);
    float* h1n     = allocf(BD);
    float* qatt    = allocf(BD);
    float* vctx    = allocf(BD);
    float* nw2     = allocf(BD * 2);
    float* gates2  = allocf((size_t)BN * 2048);
    float* lq      = allocf(BD);
    float* lattout = allocf(BD);
    float* lmha    = allocf(BD);
    float* fv      = allocf(BD * 2);
    float* r1 = allocf(BD);
    float* r2 = allocf(BD);
    float* r3 = allocf(BD);
    float* r4 = allocf(BD);
    float* ys      = allocf((size_t)BN * TSTEPS * 1024);
    float* outb    = allocf((size_t)BN * 20 * 1024);
    float* logitsb = allocf((size_t)BN * 20 * VV);

    // ---- Pack all B operands into fragment-ordered bf16 (once per call) ----
    auto pack = [&](const float* W, int Kp, int Np, long ldw,
                    int realK, int realN, bool tB) -> const __bf16* {
        __bf16* p = allocb((size_t)Kp * Np);
        pack_b_kernel<<<(unsigned)(((size_t)Kp * Np) / 256), 256, 0, stream>>>(
            W, p, Np, ldw, realK, realN, tB ? 1 : 0);
        return p;
    };
    const __bf16* p_dWq    = pack(dWq,    512, 512, 512, 512, 512, false);
    const __bf16* p_dWk    = pack(dWk,    512, 512, 512, 512, 512, false);
    const __bf16* p_dWv    = pack(dWv,    512, 512, 512, 512, 512, false);
    const __bf16* p_dWo    = pack(dWo,    512, 512, 512, 512, 512, false);
    const __bf16* p_lWq    = pack(lWq,    512, 512, 512, 512, 512, false);
    const __bf16* p_lWk    = pack(lWk,    512, 512, 512, 512, 512, false);
    const __bf16* p_lWv    = pack(lWv,    512, 512, 512, 512, 512, false);
    const __bf16* p_lWo    = pack(lWo,    512, 512, 512, 512, 512, false);
    const __bf16* p_attw1  = pack(att_w1, 512, 512, 512, 512, 512, false);
    const __bf16* p_attw2  = pack(att_w2, 512, 512, 512, 512, 512, false);
    const __bf16* p_ihw    = pack(init_h_w, 512, 512, 512, 512, 512, false);
    const __bf16* p_icw    = pack(init_c_w, 512, 512, 512, 512, 512, false);
    const __bf16* p_l1ih   = pack(l1_wih, 1024, 2048, 1024, 1024, 2048, true);
    const __bf16* p_l1hh   = pack(l1_whh,  512, 2048,  512,  512, 2048, true);
    const __bf16* p_l2ih   = pack(l2_wih, 1024, 2048, 1024, 1024, 2048, true);
    const __bf16* p_l2hh   = pack(l2_whh,  512, 2048,  512,  512, 2048, true);
    const __bf16* p_lp     = pack(l_param, 512, LP_PAD, 100, 512, 100, false);
    const __bf16* p_lpT    = pack(l_param, LP_PAD, 512, 100, 100, 512, true);
    const __bf16* p_linw   = pack(lin_w,  1024, 512, 512, 1024, 512, false);
    const __bf16* p_lin1w  = pack(lin1_w, 1024, 512, 512, 1024, 512, false);
    const __bf16* p_lin2w  = pack(lin2_w,  512, 512, 512,  512, 512, false);
    const __bf16* p_lin3w  = pack(lin3_w,  512, 512, 512,  512, 512, false);
    const __bf16* p_vocab  = pack(vocab_w, 1024, VV, VV, 1024, VV, false);

    const int EW = (BN * DDIM) / 256;   // 1024 blocks for B*D elementwise grids

    // ---- Step-invariant projections (hoisted out of the scan) ----
    launch_gemm(stream, sg,  DDIM, p_dWk,   nullptr, sgK,   BN * SSGD, HDK, DDIM, 0, false);
    launch_gemm(stream, sg,  DDIM, p_dWv,   nullptr, sgV,   BN * SSGD, HDK, DDIM, 0, false);
    launch_gemm(stream, enc, DDIM, p_lWk,   nullptr, encK,  BN * SDIM, HDK, DDIM, 0, false);
    launch_gemm(stream, enc, DDIM, p_lWv,   nullptr, encV,  BN * SDIM, HDK, DDIM, 0, false);
    launch_gemm(stream, enc, DDIM, p_attw1, nullptr, encW1, BN * SDIM, DDIM, DDIM, 0, false);

    // ---- Initial carry ----
    launch_gemm(stream, mean, DDIM, p_ihw, init_h_b, hA[0], BN, DDIM, DDIM, 1, false);
    dup_kernel<<<EW, 256, 0, stream>>>(hA[0], hA[1], hA[2]);
    launch_gemm(stream, mean, DDIM, p_icw, init_c_b, cA[0], BN, DDIM, DDIM, 1, false);
    dup_kernel<<<EW, 256, 0, stream>>>(cA[0], cA[1], nullptr);

    float* hc[3] = { hA[0], hA[1], hA[2] };
    float* hn[3] = { hB[0], hB[1], hB[2] };
    float* cc[2] = { cA[0], cA[1] };
    float* cn[2] = { cB[0], cB[1] };

    const dim3 mhaGrid(BN * HH / 8), mhaBlk(32, 8);

    // ---- Sequential scan over 19 time steps ----
    for (int t = 0; t < TSTEPS; ++t) {
        const float* x = capv + (size_t)t * DDIM;           // [B] rows, stride L*D

        // decoder memory-MHA over scene graph + residual + LN
        launch_gemm(stream, x, (long)LDIM * DDIM, p_dWq, nullptr, qbuf, BN, HDK, DDIM, 0, false);
        mha_kernel<<<mhaGrid, mhaBlk, 0, stream>>>(qbuf, sgK, sgV, d_mk, d_mv, sgmask2, attout, SSGD);
        launch_gemm(stream, attout, HDK, p_dWo, nullptr, mhaout, BN, DDIM, HDK, 0, false);
        ln_res_kernel<<<BN, 256, 0, stream>>>(mhaout, x, (long)LDIM * DDIM, x1);

        // LSTM1 on concat((mean+h2+h3)/3, x1)
        nwbuild_kernel<<<EW, 256, 0, stream>>>(mean, hc[1], hc[2], x1, nw);
        launch_gemm(stream, nw,    1024, p_l1ih, nullptr, gates, BN, 2048, 1024, 0, false);
        launch_gemm(stream, hc[0], DDIM, p_l1hh, l1_b,    gates, BN, 2048, DDIM, 0, true);
        lstm_ew_kernel<<<EW, 256, 0, stream>>>(gates, cc[0], hn[0], cn[0]);

        // h1n = relu(softmax(h1 @ l_param) @ l_param.T) + h1   (100 padded to 128)
        launch_gemm(stream, hn[0], DDIM, p_lp, nullptr, t100, BN, LP_PAD, DDIM, 0, false);
        softmax100_kernel<<<BN, 32, 0, stream>>>(t100);
        launch_gemm(stream, t100, LP_PAD, p_lpT, nullptr, h1npre, BN, DDIM, LP_PAD, 0, false);
        reluadd_kernel<<<EW, 256, 0, stream>>>(h1npre, hn[0], h1n);

        // Bahdanau attention
        launch_gemm(stream, h1n, DDIM, p_attw2, nullptr, qatt, BN, DDIM, DDIM, 0, false);
        addatt_kernel<<<BN, 256, 0, stream>>>(encW1, qatt, att_v, enc, sgmask, vctx);

        // LSTM2 on concat(h1n, v)
        concat_kernel<<<EW, 256, 0, stream>>>(h1n, nullptr, vctx, nw2);
        launch_gemm(stream, nw2,   1024, p_l2ih, nullptr, gates2, BN, 2048, 1024, 0, false);
        launch_gemm(stream, hc[1], DDIM, p_l2hh, l2_b,    gates2, BN, 2048, DDIM, 0, true);
        lstm_ew_kernel<<<EW, 256, 0, stream>>>(gates2, cc[1], hn[1], cn[1]);

        // second memory-MHA over encoder + gated output
        launch_gemm(stream, h1n, DDIM, p_lWq, nullptr, lq, BN, HDK, DDIM, 0, false);
        mha_kernel<<<mhaGrid, mhaBlk, 0, stream>>>(lq, encK, encV, l_mk, l_mv, sgmask, lattout, SDIM);
        launch_gemm(stream, lattout, HDK, p_lWo, nullptr, lmha, BN, DDIM, HDK, 0, false);
        concat_kernel<<<EW, 256, 0, stream>>>(lmha, h1n, vctx, fv);   // fv = [lmha+h1n, v]
        launch_gemm(stream, fv,   1024, p_linw,  lin_b,  r1, BN, DDIM, 1024, 1, false);
        launch_gemm(stream, qatt, DDIM, p_lin2w, lin2_b, r2, BN, DDIM, DDIM, 1, false);
        launch_gemm(stream, fv,   1024, p_lin1w, lin1_b, r3, BN, DDIM, 1024, 1, false);
        launch_gemm(stream, qatt, DDIM, p_lin3w, lin3_b, r4, BN, DDIM, DDIM, 1, false);
        combine_kernel<<<EW, 256, 0, stream>>>(r1, r2, r3, r4, hn[1], hn[2], ys, t);

        // swap carry banks
        for (int i = 0; i < 3; ++i) { float* tmp = hc[i]; hc[i] = hn[i]; hn[i] = tmp; }
        for (int i = 0; i < 2; ++i) { float* tmp = cc[i]; cc[i] = cn[i]; cn[i] = tmp; }
    }

    // ---- Epilogue: BOS row + vocab projection + log_softmax + argmax ----
    fillout_kernel<<<(BN * 20 * 1024) / 256, 256, 0, stream>>>(ys, outb);
    launch_gemm(stream, outb, 1024, p_vocab, nullptr, logitsb, BN * 20, VV, 1024, 1, false);
    logsoftmax_kernel<<<BN * 20, 256, 0, stream>>>(logitsb, (float*)d_out);
    argmax_kernel<<<(BN * VV) / 256, 256, 0, stream>>>(
        logitsb, (int*)((float*)d_out + (size_t)BN * 20 * VV));
}
